// MultiHeadLatentAttention_59236188947241
// MI455X (gfx1250) — compile-verified, matrix operand order checked by reading, not run on hardware
//
#include <hip/hip_runtime.h>
#include <hip/hip_bf16.h>
#include <math.h>

typedef __bf16 bf16;
typedef __attribute__((ext_vector_type(16))) __bf16 v16bf;
typedef __attribute__((ext_vector_type(4)))  __bf16 v4bf;
typedef __attribute__((ext_vector_type(8)))  float  v8f;

#define SEQ 2048
#define HID 4096
#define NH  32
#define HD  128

#if __has_builtin(__builtin_amdgcn_global_load_async_to_lds_b128) && \
    __has_builtin(__builtin_amdgcn_s_wait_asynccnt)
#define HAVE_ASYNC_LDS 1
typedef int i32x4 __attribute__((vector_size(16)));
typedef __attribute__((address_space(1))) i32x4 gvec;   // global int4
typedef __attribute__((address_space(3))) i32x4 lvec;   // LDS int4
#else
#define HAVE_ASYNC_LDS 0
#endif

__device__ __forceinline__ v8f wmma_bf16(v16bf a, v16bf b, v8f c) {
  // D = A(16x32) * B(32x16) + C, fp32 accumulate
  return __builtin_amdgcn_wmma_f32_16x16x32_bf16(false, a, false, b, (short)0, c, false, false);
}

// ---------------- fp32 -> bf16 conversion (vectorized) ----------------
__global__ void cvt_f32_to_bf16(const float* __restrict__ src, bf16* __restrict__ dst, int n4) {
  int i = blockIdx.x * blockDim.x + threadIdx.x;
  if (i >= n4) return;
  float4 v = ((const float4*)src)[i];
  v4bf o = { (bf16)v.x, (bf16)v.y, (bf16)v.z, (bf16)v.w };
  ((v4bf*)dst)[i] = o;
}

// ---------------- NT GEMM: C[M,N] = A[M,K] * B[N,K]^T --------------------
// Block tile 128x128, 4 waves (2x2), each wave 64x64 = 4x4 WMMA tiles.
// Both A and B fragments are contiguous 32-byte rows -> direct global loads.
__global__ __launch_bounds__(128, 1)
void gemm_nt_bf16(const bf16* __restrict__ A, const bf16* __restrict__ B,
                  bf16* __restrict__ Cb, float* __restrict__ Cf,
                  int M, int N, int K) {
  const int lane = threadIdx.x & 31;
  const int wave = threadIdx.x >> 5;
  const int wm = wave >> 1;
  const int wn = wave & 1;
  const int m0 = blockIdx.y * 128 + wm * 64;
  const int n0 = blockIdx.x * 128 + wn * 64;
  const int r  = lane & 15;   // row within 16-tile
  const int h  = lane >> 4;   // which 16-wide K half

  v8f acc[4][4];
#pragma unroll
  for (int i = 0; i < 4; ++i)
#pragma unroll
    for (int j = 0; j < 4; ++j)
#pragma unroll
      for (int e = 0; e < 8; ++e) acc[i][j][e] = 0.f;

  const bf16* arow[4];
  const bf16* brow[4];
#pragma unroll
  for (int i = 0; i < 4; ++i) arow[i] = A + (size_t)(m0 + i * 16 + r) * K + h * 16;
#pragma unroll
  for (int j = 0; j < 4; ++j) brow[j] = B + (size_t)(n0 + j * 16 + r) * K + h * 16;

  for (int k = 0; k < K; k += 32) {
    v16bf af[4], bfm[4];
#pragma unroll
    for (int i = 0; i < 4; ++i) af[i]  = *(const v16bf*)(arow[i] + k);
#pragma unroll
    for (int j = 0; j < 4; ++j) bfm[j] = *(const v16bf*)(brow[j] + k);
    __builtin_prefetch(arow[0] + k + 256, 0, 3);
    __builtin_prefetch(brow[0] + k + 256, 0, 3);
#pragma unroll
    for (int i = 0; i < 4; ++i)
#pragma unroll
      for (int j = 0; j < 4; ++j)
        acc[i][j] = wmma_bf16(af[i], bfm[j], acc[i][j]);
  }

  // C/D layout: VGPR e, lanes 0-15: M=e, N=lane; lanes 16-31: M=e+8, N=lane-16
#pragma unroll
  for (int i = 0; i < 4; ++i) {
#pragma unroll
    for (int j = 0; j < 4; ++j) {
      const int col = n0 + j * 16 + r;
#pragma unroll
      for (int e = 0; e < 8; ++e) {
        const int row = m0 + i * 16 + e + 8 * h;
        if (Cf) Cf[(size_t)row * N + col] = acc[i][j][e];
        else    Cb[(size_t)row * N + col] = (bf16)acc[i][j][e];
      }
    }
  }
}

// ---------------- RoPE on K (in place, bf16) ----------------
__global__ void rope_kernel(bf16* __restrict__ Kb) {
  int idx = blockIdx.x * blockDim.x + threadIdx.x;   // SEQ*NH*64 threads
  int d    = idx & 63;
  int t    = idx >> 6;
  int head = t & (NH - 1);
  int s    = t >> 5;
  if (s >= SEQ) return;
  // inv_freq[d] = 10000^(-d/64) = exp(-d/64 * ln(10000))
  float invf = __expf(-(float)d * (9.210340371976184f / 64.0f));
  float ang  = (float)s * invf;
  float sn, cs;
  __sincosf(ang, &sn, &cs);
  size_t base = (size_t)s * HID + (size_t)head * HD;
  float x1 = (float)Kb[base + d];
  float x2 = (float)Kb[base + d + 64];
  Kb[base + d]      = (bf16)(x1 * cs - x2 * sn);
  Kb[base + d + 64] = (bf16)(x2 * cs + x1 * sn);
}

// ---------------- V transpose: [S, nh*hd] -> [nh, hd, S] ----------------
__global__ void transpose_v(const bf16* __restrict__ V, bf16* __restrict__ Vt) {
  int idx = blockIdx.x * blockDim.x + threadIdx.x;   // SEQ*HID threads
  int col = idx & (HID - 1);
  int s   = idx >> 12;
  if (s >= SEQ) return;
  int head = col >> 7;
  int d    = col & (HD - 1);
  Vt[((size_t)head * HD + d) * SEQ + s] = V[(size_t)s * HID + col];
}

// ---------------- Flash attention, online softmax ----------------
// Grid: (SEQ/64, NH), 128 threads = 4 waves; wave handles 16 q-rows x full hd.
// K/V blocks double-buffered in LDS: CDNA5 GLOBAL_LOAD_ASYNC_TO_LDS_B128
// (ASYNCcnt) prefetches block j+1 while block j is consumed by WMMA+softmax.
__global__ __launch_bounds__(128, 1)
void flash_attn(const bf16* __restrict__ Q, const bf16* __restrict__ Kc,
                const bf16* __restrict__ Vt, const int* __restrict__ mask,
                bf16* __restrict__ O, float scale) {
  const int head = blockIdx.y;
  const int tid  = threadIdx.x;
  const int lane = tid & 31;
  const int wave = tid >> 5;
  const int q0   = blockIdx.x * 64 + wave * 16;
  const int r    = lane & 15;
  const int h    = lane >> 4;

  __shared__ __align__(64) bf16 kbuf[2][64][128];   // [buf][key][d]   32 KB
  __shared__ __align__(64) bf16 vbuf[2][128][64];   // [buf][d][key]   32 KB
  __shared__ __align__(64) bf16 pbuf[4][16][64];    // per-wave P       8 KB

  // stage K block [64 keys][128 d] and V block [128 d][64 keys] into buf.
  // 1024 chunks of 16 B each per matrix; 8 chunks per thread.
  auto stage = [&](int j0, int buf) {
    bf16* kflat = &kbuf[buf][0][0];
    bf16* vflat = &vbuf[buf][0][0];
#pragma unroll
    for (int i = 0; i < 8; ++i) {
      const int ek = (i * 128 + tid) * 8;             // element index of chunk
      const bf16* gk = Kc + (size_t)(j0 + (ek >> 7)) * HID + (size_t)head * HD + (ek & 127);
      const bf16* gv = Vt + ((size_t)head * HD + (ek >> 6)) * SEQ + j0 + (ek & 63);
#if HAVE_ASYNC_LDS
      __builtin_amdgcn_global_load_async_to_lds_b128((gvec*)gk, (lvec*)(kflat + ek), 0, 0);
      __builtin_amdgcn_global_load_async_to_lds_b128((gvec*)gv, (lvec*)(vflat + ek), 0, 0);
#else
      *(uint4*)(kflat + ek) = *(const uint4*)gk;
      *(uint4*)(vflat + ek) = *(const uint4*)gv;
#endif
    }
  };

  // Q A-fragments: 4 k-steps of 32 covering hd=128
  v16bf qf[4];
  const bf16* qrow = Q + (size_t)(q0 + r) * HID + (size_t)head * HD + h * 16;
#pragma unroll
  for (int kk = 0; kk < 4; ++kk) qf[kk] = *(const v16bf*)(qrow + kk * 32);

  v8f oacc[8];
#pragma unroll
  for (int dt = 0; dt < 8; ++dt)
#pragma unroll
    for (int e = 0; e < 8; ++e) oacc[dt][e] = 0.f;

  float m_i[8], l_i[8];
#pragma unroll
  for (int e = 0; e < 8; ++e) { m_i[e] = -1e30f; l_i[e] = 0.f; }

  // prologue: fill buffer 0
  stage(0, 0);
#if HAVE_ASYNC_LDS
  __builtin_amdgcn_s_wait_asynccnt(0);
#endif
  __syncthreads();

  for (int j0 = 0; j0 < SEQ; j0 += 64) {
    const int cur = (j0 >> 6) & 1;

    // kick off async staging of the next block (overlaps with compute below)
    if (j0 + 64 < SEQ) stage(j0 + 64, cur ^ 1);

    // ---- scores: 4 tiles of 16 keys (Q 16xhd @ K^T), K frags from LDS ----
    v8f st[4];
#pragma unroll
    for (int t = 0; t < 4; ++t) {
      v8f a;
#pragma unroll
      for (int e = 0; e < 8; ++e) a[e] = 0.f;
#pragma unroll
      for (int kk = 0; kk < 4; ++kk) {
        v16bf kf = *(const v16bf*)&kbuf[cur][t * 16 + r][kk * 32 + h * 16];
        a = wmma_bf16(qf[kk], kf, a);
      }
      int mv = mask[j0 + t * 16 + r];   // key column for this lane
#pragma unroll
      for (int e = 0; e < 8; ++e)
        st[t][e] = (mv == 0) ? -3.0e38f : a[e] * scale;
    }

    // ---- block row-max (reduce across the 16 lanes that hold a row) ----
    float bm[8];
#pragma unroll
    for (int e = 0; e < 8; ++e)
      bm[e] = fmaxf(fmaxf(st[0][e], st[1][e]), fmaxf(st[2][e], st[3][e]));
#pragma unroll
    for (int off = 1; off < 16; off <<= 1) {
#pragma unroll
      for (int e = 0; e < 8; ++e)
        bm[e] = fmaxf(bm[e], __shfl_xor(bm[e], off, 32));
    }

    float alpha[8];
#pragma unroll
    for (int e = 0; e < 8; ++e) {
      float mn = fmaxf(m_i[e], bm[e]);
      alpha[e] = __expf(m_i[e] - mn);
      m_i[e]   = mn;
    }

    // ---- p = exp(s - m), row sums ----
    float ls[8];
#pragma unroll
    for (int e = 0; e < 8; ++e) ls[e] = 0.f;
#pragma unroll
    for (int t = 0; t < 4; ++t)
#pragma unroll
      for (int e = 0; e < 8; ++e) {
        float p = __expf(st[t][e] - m_i[e]);
        st[t][e] = p;
        ls[e] += p;
      }
#pragma unroll
    for (int off = 1; off < 16; off <<= 1) {
#pragma unroll
      for (int e = 0; e < 8; ++e) ls[e] += __shfl_xor(ls[e], off, 32);
    }
#pragma unroll
    for (int e = 0; e < 8; ++e) l_i[e] = l_i[e] * alpha[e] + ls[e];

    // ---- rescale running O ----
#pragma unroll
    for (int dt = 0; dt < 8; ++dt)
#pragma unroll
      for (int e = 0; e < 8; ++e) oacc[dt][e] *= alpha[e];

    // ---- C-layout -> A-layout shuffle of P through LDS ----
#pragma unroll
    for (int t = 0; t < 4; ++t)
#pragma unroll
      for (int e = 0; e < 8; ++e)
        pbuf[wave][e + 8 * h][t * 16 + r] = (bf16)st[t][e];
    __syncthreads();

    v16bf pf[2];
#pragma unroll
    for (int kk = 0; kk < 2; ++kk)
      pf[kk] = *(const v16bf*)&pbuf[wave][r][kk * 32 + h * 16];

    // ---- O += P(16x64) @ V(64xhd), V frags from LDS ----
#pragma unroll
    for (int dt = 0; dt < 8; ++dt) {
#pragma unroll
      for (int kk = 0; kk < 2; ++kk) {
        v16bf vf = *(const v16bf*)&vbuf[cur][dt * 16 + r][kk * 32 + h * 16];
        oacc[dt] = wmma_bf16(pf[kk], vf, oacc[dt]);
      }
    }

    // retire async staging, publish next buffer, release current one
#if HAVE_ASYNC_LDS
    __builtin_amdgcn_s_wait_asynccnt(0);
#endif
    __syncthreads();
  }

  // ---- normalize and write O[s, head*hd + d] (bf16) ----
#pragma unroll
  for (int dt = 0; dt < 8; ++dt) {
    const int col = head * HD + dt * 16 + r;
#pragma unroll
    for (int e = 0; e < 8; ++e) {
      const int row = q0 + e + 8 * h;
      O[(size_t)row * HID + col] = (bf16)(oacc[dt][e] / l_i[e]);
    }
  }
}

// ---------------- launcher ----------------
extern "C" void kernel_launch(void* const* d_in, const int* in_sizes, int n_in,
                              void* d_out, int out_size, void* d_ws, size_t ws_size,
                              hipStream_t stream) {
  (void)in_sizes; (void)n_in; (void)out_size; (void)ws_size;
  const float* hs  = (const float*)d_in[0];
  const int*   msk = (const int*)d_in[1];
  const float* Wq  = (const float*)d_in[2];
  const float* Wk  = (const float*)d_in[3];
  const float* Wv  = (const float*)d_in[4];
  const float* Wo  = (const float*)d_in[5];
  const float* Lq  = (const float*)d_in[6];
  float* out = (float*)d_out;

  char* ws = (char*)d_ws;
  const size_t SZ_ACT = (size_t)SEQ * HID * sizeof(bf16);   // 16 MB
  const size_t SZ_W   = (size_t)HID * HID * sizeof(bf16);   // 32 MB
  bf16* Xb = (bf16*)(ws);
  bf16* Lb = (bf16*)(ws + SZ_ACT);
  bf16* W0 = (bf16*)(ws + 2 * SZ_ACT);                // Wq, later reused for Wo
  bf16* W1 = (bf16*)(ws + 2 * SZ_ACT + SZ_W);         // Wk
  bf16* W2 = (bf16*)(ws + 2 * SZ_ACT + 2 * SZ_W);     // Wv
  bf16* Qb = (bf16*)(ws + 2 * SZ_ACT + 3 * SZ_W);
  bf16* Kb = (bf16*)(ws + 3 * SZ_ACT + 3 * SZ_W);
  bf16* Vb = (bf16*)(ws + 4 * SZ_ACT + 3 * SZ_W);
  bf16* Vt = Xb;   // X dead after V projection
  bf16* Ob = Lb;   // latents dead after Q projection

  const int nAct = SEQ * HID;   // 8,388,608
  const int nW   = HID * HID;   // 16,777,216

  cvt_f32_to_bf16<<<nAct / 4 / 256, 256, 0, stream>>>(hs, Xb, nAct / 4);
  cvt_f32_to_bf16<<<nAct / 4 / 256, 256, 0, stream>>>(Lq, Lb, nAct / 4);
  cvt_f32_to_bf16<<<nW / 4 / 256, 256, 0, stream>>>(Wq, W0, nW / 4);
  cvt_f32_to_bf16<<<nW / 4 / 256, 256, 0, stream>>>(Wk, W1, nW / 4);
  cvt_f32_to_bf16<<<nW / 4 / 256, 256, 0, stream>>>(Wv, W2, nW / 4);

  dim3 gg(HID / 128, SEQ / 128);
  gemm_nt_bf16<<<gg, 128, 0, stream>>>(Lb, W0, Qb, nullptr, SEQ, HID, HID);
  gemm_nt_bf16<<<gg, 128, 0, stream>>>(Xb, W1, Kb, nullptr, SEQ, HID, HID);
  gemm_nt_bf16<<<gg, 128, 0, stream>>>(Xb, W2, Vb, nullptr, SEQ, HID, HID);

  cvt_f32_to_bf16<<<nW / 4 / 256, 256, 0, stream>>>(Wo, W0, nW / 4);   // reuse Wq slot

  rope_kernel<<<(SEQ * NH * 64) / 256, 256, 0, stream>>>(Kb);
  transpose_v<<<nAct / 256, 256, 0, stream>>>(Vb, Vt);

  dim3 fg(SEQ / 64, NH);
  flash_attn<<<fg, 128, 0, stream>>>(Qb, Kb, Vt, msk, Ob, 0.08838834764831845f);

  gemm_nt_bf16<<<gg, 128, 0, stream>>>(Ob, W0, nullptr, out, SEQ, HID, HID);
}
